// Head_76227079569570
// MI455X (gfx1250) — matrix operand order repaired and checked by reading, not compiled, not run on hardware
//
#include <hip/hip_runtime.h>
#include <hip/hip_bf16.h>

#define B_ 8
#define T_ 4096
#define C_ 384
#define H_ 64

typedef __attribute__((ext_vector_type(16))) _Float16 v16h;
typedef __attribute__((ext_vector_type(8)))  _Float16 v8h;
typedef __attribute__((ext_vector_type(8)))  float    v8f;
typedef __attribute__((ext_vector_type(4)))  float    v4f;

// ISA 7.12.2: 16-bit A (16x32) / B (32x16) fragments: half-element h of lane l
// maps to K = (h>=8 ? 16:0) + (l>=16 ? 8:0) + (h&7)  -> two contiguous 8-runs.
__device__ __forceinline__ int kOf(int h, int lane) {
    return ((h & 8) << 1) | ((lane & 16) >> 1) | (h & 7);
}

__device__ __forceinline__ v8f wmma32f16(v16h a, v16h b, v8f c) {
    return __builtin_amdgcn_wmma_f32_16x16x32_f16(
        false, a, false, b, (short)0, c, false, false);
}

__device__ __forceinline__ v8f zero8() {
    v8f z;
#pragma unroll
    for (int i = 0; i < 8; ++i) z[i] = 0.0f;
    return z;
}

__device__ __forceinline__ v16h cat16(v8h lo, v8h hi) {
    v16h r;
#pragma unroll
    for (int i = 0; i < 8; ++i) { r[i] = lo[i]; r[8 + i] = hi[i]; }
    return r;
}

__device__ __forceinline__ v16h loadFrag16(const _Float16* p) {
    return cat16(*(const v8h*)p, *(const v8h*)(p + 16));
}

__device__ __forceinline__ float max8(v8f v) {
    const float a = fmaxf(fmaxf(v[0], v[1]), fmaxf(v[2], v[3]));
    const float b = fmaxf(fmaxf(v[4], v[5]), fmaxf(v[6], v[7]));
    return fmaxf(a, b);
}
__device__ __forceinline__ float sum8(v8f v) {
    return ((v[0] + v[1]) + (v[2] + v[3])) + ((v[4] + v[5]) + (v[6] + v[7]));
}

// 1/sqrt(H) * log2(e): folded into Q at projection-store time (base-2 softmax)
#define QSCALE 0.180336878f

// ---------------------------------------------------------------------------
// Kernel 0: one-time swizzle of Wq/Wk/Wv into f16 fragment layout.
// ---------------------------------------------------------------------------
__global__ void __launch_bounds__(128)
swizzle_w(const float* __restrict__ Wq, const float* __restrict__ Wk,
          const float* __restrict__ Wv, _Float16* __restrict__ wsw)
{
    const int tid  = blockIdx.x * 128 + threadIdx.x;   // 0..4607
    const int lane = tid & 31;
    const int frag = tid >> 5;                         // 0..143
    const int nt   = frag & 3;
    const int kc   = (frag >> 2) % 12;
    const int w    = frag / 48;
    const float* W = (w == 0) ? Wq : (w == 1) ? Wk : Wv;
    const int kb   = kc * 32;
    const int n    = lane & 15;

    v8h lo, hi;
#pragma unroll
    for (int i = 0; i < 8; ++i) {
        lo[i] = (_Float16)W[(kb + kOf(i, lane)) * H_ + nt * 16 + n];
        hi[i] = (_Float16)W[(kb + kOf(8 + i, lane)) * H_ + nt * 16 + n];
    }
    _Float16* dst = wsw + ((size_t)frag * 32 + lane) * 16;
    *(v8h*)dst       = lo;
    *(v8h*)(dst + 8) = hi;
}

// ---------------------------------------------------------------------------
// Kernel 1: transposed projection (out^T = W^T @ x^T); grid.y selects q/k/v
// so each block carries only 4 accumulator tiles and pipelines its loads.
// x re-reads hit L2 (x = 50MB, L2 = 192MB).
// ---------------------------------------------------------------------------
__global__ void __launch_bounds__(128)
proj_one(const float* __restrict__ x, const _Float16* __restrict__ wsw,
         _Float16* __restrict__ qo, _Float16* __restrict__ ko,
         _Float16* __restrict__ vt)
{
    const int w    = blockIdx.y;                 // 0=q, 1=k, 2=v
    const int lane = threadIdx.x & 31;
    const int wid  = threadIdx.x >> 5;
    const int rt   = blockIdx.x * 4 + wid;
    const int n    = lane & 15;
    const int off  = (lane & 16) >> 1;
    const int arow = rt * 16 + n;                // token handled by this lane

    const _Float16* wswm = wsw + (size_t)w * 24576;   // 48*32*16 per matrix

    v8f acc[4];
#pragma unroll
    for (int nt = 0; nt < 4; ++nt) acc[nt] = zero8();

    for (int kc = 0; kc < 12; ++kc) {
        const int kb = kc * 32;
        const float* xp = x + (size_t)arow * C_ + kb + off;
        const v4f x0 = *(const v4f*)(xp);
        const v4f x1 = *(const v4f*)(xp + 4);
        const v4f x2 = *(const v4f*)(xp + 16);
        const v4f x3 = *(const v4f*)(xp + 20);
        v16h bf[4];
#pragma unroll
        for (int nt = 0; nt < 4; ++nt)
            bf[nt] = *(const v16h*)(wswm + ((size_t)(kc * 4 + nt) * 32 + lane) * 16);
        v16h xf;                                  // x^T B-fragment
#pragma unroll
        for (int i = 0; i < 4; ++i) {
            xf[i]      = (_Float16)x0[i];
            xf[4 + i]  = (_Float16)x1[i];
            xf[8 + i]  = (_Float16)x2[i];
            xf[12 + i] = (_Float16)x3[i];
        }
#pragma unroll
        for (int nt = 0; nt < 4; ++nt)
            acc[nt] = wmma32f16(bf[nt], xf, acc[nt]);   // W^T as A, x^T as B
    }

    const int g  = (lane >> 4) << 3;             // head-col base for this lane
    const int bb = (rt * 16) >> 12;
    const int t0 = (rt * 16) & (T_ - 1);
    if (w == 0) {                                // q: pre-scaled, packed rows
#pragma unroll
        for (int nt = 0; nt < 4; ++nt) {
            v8h hv;
#pragma unroll
            for (int r = 0; r < 8; ++r) hv[r] = (_Float16)(acc[nt][r] * QSCALE);
            *(v8h*)(qo + (size_t)arow * H_ + nt * 16 + g) = hv;
        }
    } else if (w == 1) {                         // k: packed rows
#pragma unroll
        for (int nt = 0; nt < 4; ++nt) {
            v8h hv;
#pragma unroll
            for (int r = 0; r < 8; ++r) hv[r] = (_Float16)acc[nt][r];
            *(v8h*)(ko + (size_t)arow * H_ + nt * 16 + g) = hv;
        }
    } else {                                     // v: transposed [B][H][T]
#pragma unroll
        for (int nt = 0; nt < 4; ++nt)
#pragma unroll
            for (int r = 0; r < 8; ++r)
                vt[((size_t)bb * H_ + nt * 16 + g + r) * T_ + t0 + n] =
                    (_Float16)acc[nt][r];
    }
}

// ---------------------------------------------------------------------------
// Per-tile flash attention body (register-resident):
//   S^T = K @ Q^T ; P^T fragment built in registers ; O^T = V^T @ P^T.
// Conditional accumulator rescale (running max is a record process).
// ---------------------------------------------------------------------------
__device__ __forceinline__ void attn_tile(
    int qt, int lane, const _Float16* __restrict__ q,
    const _Float16* __restrict__ k, const _Float16* __restrict__ vt,
    float* __restrict__ out)
{
    const int qrow0 = qt * 16;
    const int bb    = qrow0 >> 12;
    const int t0    = qrow0 & (T_ - 1);
    const int n     = lane & 15;
    const int off   = (lane & 16) >> 1;
    const int g     = (lane >> 4) << 3;

    // Q fragments (B operand of S^T = K @ Q^T); query of this lane = t0 + n
    const _Float16* qp = q + ((size_t)qrow0 + n) * H_ + off;
    const v16h qa0 = loadFrag16(qp);
    const v16h qa1 = loadFrag16(qp + 32);

    float mRow = -__builtin_inff();
    float lRow = 0.0f;
    v8f acc[4];                       // O^T tiles: (head-col r+g, query n)
#pragma unroll
    for (int nt = 0; nt < 4; ++nt) acc[nt] = zero8();

    const _Float16* kbase = k + (size_t)bb * T_ * H_;
    const _Float16* vbb   = vt + (size_t)bb * H_ * T_;
    const int npairs = (t0 + 16 + 31) >> 5;   // last step is the diagonal one

    for (int j = 0; j < npairs; ++j) {
        const int kb0 = j * 32;

        // K tile A-fragments + V^T A-fragments, all b128 loads
        const _Float16* kp0 = kbase + (size_t)(kb0 + n) * H_ + off;
        const _Float16* kp1 = kp0 + (size_t)16 * H_;
        const v16h ka00 = loadFrag16(kp0);
        const v16h ka01 = loadFrag16(kp0 + 32);
        const v16h ka10 = loadFrag16(kp1);
        const v16h ka11 = loadFrag16(kp1 + 32);
        v16h vf[4];
#pragma unroll
        for (int nt = 0; nt < 4; ++nt)
            vf[nt] = loadFrag16(vbb + (size_t)(nt * 16 + n) * T_ + kb0 + off);

        // S^T tiles: lane holds query n, keys kb0+g+r / kb0+16+g+r
        v8f st0 = zero8(), st1 = zero8();
        st0 = wmma32f16(ka00, qa0, st0);  st0 = wmma32f16(ka01, qa1, st0);
        st1 = wmma32f16(ka10, qa0, st1);  st1 = wmma32f16(ka11, qa1, st1);

        if (j == npairs - 1) {             // causal mask, diagonal step only
            const int lim = t0 + n - kb0;
#pragma unroll
            for (int r = 0; r < 8; ++r) {
                st0[r] = (g + r      <= lim) ? st0[r] : -__builtin_inff();
                st1[r] = (g + r + 16 <= lim) ? st1[r] : -__builtin_inff();
            }
        }

        float mx = fmaxf(max8(st0), max8(st1));
        mx = fmaxf(mx, __shfl_xor(mx, 16, 32));

        // rescale only when the wave-wide running max grew (rare)
        if (__any(mx > mRow)) {
            const float mnew  = fmaxf(mRow, mx);
            const float alpha = __builtin_exp2f(mRow - mnew);
            lRow *= alpha;
#pragma unroll
            for (int nt = 0; nt < 4; ++nt)
#pragma unroll
                for (int r = 0; r < 8; ++r) acc[nt][r] *= alpha;
            mRow = mnew;
        }

        v8f p0, p1;
#pragma unroll
        for (int r = 0; r < 8; ++r) {
            p0[r] = __builtin_exp2f(st0[r] - mRow);
            p1[r] = __builtin_exp2f(st1[r] - mRow);
        }
        float ps = sum8(p0) + sum8(p1);
        ps += __shfl_xor(ps, 16, 32);
        lRow += ps;

        // P^T B-fragment is (p0,p1) packed to f16 in registers (no LDS)
        v8h phl, phh;
#pragma unroll
        for (int r = 0; r < 8; ++r) { phl[r] = (_Float16)p0[r]; phh[r] = (_Float16)p1[r]; }
        const v16h pB = cat16(phl, phh);

        // O^T += V^T @ P^T
#pragma unroll
        for (int nt = 0; nt < 4; ++nt)
            acc[nt] = wmma32f16(vf[nt], pB, acc[nt]);
    }

    // normalize (lane-local) and write: 8 consecutive head-cols per lane
    const float linv = 1.0f / lRow;
    float* op = out + ((size_t)qrow0 + n) * H_ + g;
#pragma unroll
    for (int nt = 0; nt < 4; ++nt) {
        v4f olo, ohi;
#pragma unroll
        for (int i = 0; i < 4; ++i) {
            olo[i] = acc[nt][i]     * linv;
            ohi[i] = acc[nt][4 + i] * linv;
        }
        *(v4f*)(op + nt * 16)     = olo;
        *(v4f*)(op + nt * 16 + 4) = ohi;
    }
}

// ---------------------------------------------------------------------------
// Kernel 2: each wave handles tiles qt and (2047-qt): within a batch the
// causal trip counts sum to a constant (t0_a + t0_b = 4080), so every wave
// does ~129 steps -> no triangular load-imbalance tail.
// ---------------------------------------------------------------------------
__global__ void __launch_bounds__(128)
flash_attn(const _Float16* __restrict__ q, const _Float16* __restrict__ k,
           const _Float16* __restrict__ vt, float* __restrict__ out)
{
    const int lane = threadIdx.x & 31;
    const int wid  = threadIdx.x >> 5;
    const int qt   = blockIdx.x * 4 + wid;            // 0..1023
    attn_tile(qt,                        lane, q, k, vt, out);
    attn_tile((B_ * T_ / 16 - 1) - qt,   lane, q, k, vt, out);
}

// ---------------------------------------------------------------------------
extern "C" void kernel_launch(void* const* d_in, const int* in_sizes, int n_in,
                              void* d_out, int out_size, void* d_ws, size_t ws_size,
                              hipStream_t stream) {
    const float* x  = (const float*)d_in[0];
    const float* Wq = (const float*)d_in[1];
    const float* Wk = (const float*)d_in[2];
    const float* Wv = (const float*)d_in[3];

    const size_t NQKV = (size_t)B_ * T_ * H_;    // 2,097,152 elements
    _Float16* q16  = (_Float16*)d_ws;
    _Float16* k16  = q16 + NQKV;
    _Float16* vt16 = k16 + NQKV;                 // V transposed [B][H][T]
    _Float16* wsw  = vt16 + NQKV;                // 73,728 swizzled W halves

    swizzle_w<<<36, 128, 0, stream>>>(Wq, Wk, Wv, wsw);
    proj_one<<<dim3(512, 3), 128, 0, stream>>>(x, wsw, q16, k16, vt16);
    flash_attn<<<256, 128, 0, stream>>>(q16, k16, vt16, (float*)d_out);
}
